// ProteinMPNN_42176578846969
// MI455X (gfx1250) — matrix operand order repaired and checked by reading, not compiled, not run on hardware
//
#include <hip/hip_runtime.h>
#include <hip/hip_bf16.h>
#include <math.h>

// ---------------- problem constants ----------------
#define Hc   128
#define CEc  128
#define HCc  256      // H + CE
#define Kc   48
#define Bc   4
#define Nc   2048
#define FFc  512
#define NPAIR (Bc * Nc)          // 8192
#define INV_SCALE (1.0f / 30.0f)

// ---------------- vector types ----------------
typedef __attribute__((ext_vector_type(16))) __bf16 v16bf;
typedef __attribute__((ext_vector_type(8)))  __bf16 bf16x8;
typedef __attribute__((ext_vector_type(4)))  __bf16 bf16x4;
typedef __attribute__((ext_vector_type(8)))  float  v8f;
typedef __attribute__((ext_vector_type(4)))  float  f4;
typedef __attribute__((ext_vector_type(4)))  unsigned int u32x4;
typedef __attribute__((ext_vector_type(8)))  int    i32x8;
typedef __attribute__((ext_vector_type(4)))  int    i32x4;

// ---------------- WMMA helpers ----------------
// A-fragment (16x32 bf16): lane l holds row m=l&15; 16 contiguous K starting
// at k0 + (l>>4)*16.
// => base at row*ld + k0 + (l>>4)*16, gap 8.
// B-fragment (32x16 bf16): lane l holds col n=l&15; 16 contiguous K starting
// at k0 + (l>>4)*16.  => base at col*ld + k0 + (l>>4)*16, gap 8.
__device__ __forceinline__ v16bf ld_frag(const __bf16* p, int gap) {
  bf16x8 lo = *(const bf16x8*)p;
  bf16x8 hi = *(const bf16x8*)(p + gap);
  v16bf r;
#pragma unroll
  for (int i = 0; i < 8; ++i) { r[i] = lo[i]; r[i + 8] = hi[i]; }
  return r;
}

__device__ __forceinline__ v8f wmma_bf16(v16bf a, v16bf b, v8f c) {
  return __builtin_amdgcn_wmma_f32_16x16x32_bf16(
      false, a, false, b, (short)0, c, false, false);
}

// gelu approximate=True (tanh form), tanh via fast exp
__device__ __forceinline__ float gelu_tanh(float x) {
  float u = 0.7978845608f * (x + 0.044715f * x * x * x);
  float e = __expf(2.0f * u);
  float t = __fdividef(e - 1.0f, e + 1.0f);
  return 0.5f * x * (1.0f + t);
}
// gelu approximate=False (erf form)
__device__ __forceinline__ float gelu_erf(float x) {
  return 0.5f * x * (1.0f + erff(x * 0.70710678f));
}

// ---------------- TDM: async 2D f32 tile load global->LDS ----------------
// D# per CDNA5 ISA ch.8: group0 = {count/flags, lds_addr, global_addr, type},
// group1 = {data_size + dims/strides}, groups 2/3 unused (zeros).
// This toolchain exposes the 6-arg builtin:
//   (u32x4 g0, i32x8 g1, i32x4 g2, i32x4 g3, i32x8 extra, i32 cpol)
__device__ __forceinline__ void tdm_load_f32_tile(const float* gsrc, void* lds_dst,
                                                  unsigned rows, unsigned cols,
                                                  unsigned stride) {
  unsigned long long ga = (unsigned long long)(uintptr_t)gsrc;
  unsigned lds_off = (unsigned)(uintptr_t)lds_dst;   // low 32 bits = LDS byte addr
  u32x4 g0;
  g0[0] = 1u;                                        // count=1 (valid), load, user
  g0[1] = lds_off;                                   // lds_addr
  g0[2] = (unsigned)(ga & 0xffffffffu);              // global_addr[31:0]
  g0[3] = (unsigned)((ga >> 32) & 0x1ffffffu)        // global_addr[56:32]
        | (2u << 30);                                // type=2 ("image")
  i32x8 g1;
  g1[0] = (int)(2u << 16);                           // data_size=2 -> 4 bytes
  g1[1] = (int)((cols & 0xffffu) << 16);             // tensor_dim0[15:0]
  g1[2] = (int)((cols >> 16) | ((rows & 0xffffu) << 16));   // dim0 hi | dim1 lo
  g1[3] = (int)((rows >> 16) | ((cols & 0xffffu) << 16));   // dim1 hi | tile_dim0
  g1[4] = (int)(rows & 0xffffu);                     // tile_dim1 | tile_dim2=0
  g1[5] = (int)stride;                               // tensor_dim0_stride[31:0]
  g1[6] = 0;                                         // stride0 hi | dim1_stride lo
  g1[7] = 0;
  i32x4 gz4 = {0, 0, 0, 0};
  i32x8 gz8 = {0, 0, 0, 0, 0, 0, 0, 0};
  __builtin_amdgcn_tensor_load_to_lds(g0, g1, gz4, gz4, gz8, 0);
}

// =====================================================================
// Kernel 1: edge-MLP  m = (gelu(gelu([h_nb|h_E]W1+b1)W2+b2))W3+b3,
//           masked sum over K, /30, + h_V  -> hPre (f32)
// LDS layout (bytes):
#define K1_OFF_W1T   0                         // 128 x 256 bf16  = 65536
#define K1_OFF_W2T   (K1_OFF_W1T + 65536)      // 128 x 128 bf16  = 32768
#define K1_OFF_W3T   (K1_OFF_W2T + 32768)      // 128 x 128 bf16  = 32768
#define K1_OFF_A0    (K1_OFF_W3T + 32768)      // 48 x 264 bf16   = 25344
#define K1_OFF_MID   (K1_OFF_A0 + 25344)       // 48 x 136 bf16   = 13056
#define K1_OFF_MID2  (K1_OFF_MID + 13056)      // 48 x 136 bf16   = 13056
#define K1_OFF_IDX   (K1_OFF_MID2 + 13056)     // 48 int          = 192
#define K1_OFF_MSK   (K1_OFF_IDX + 192)        // 48 float        = 192
#define K1_OFF_STG   (K1_OFF_MSK + 192)        // 48 x 128 f32    = 24576 (TDM dst)
#define K1_SMEM      (K1_OFF_STG + 24576)      // 207488 B
#define LDA0 264
#define LDM  136
// =====================================================================
__global__ __launch_bounds__(256, 1)
void enc_edge_mlp(const float* __restrict__ hV, const float* __restrict__ hE,
                  const int* __restrict__ Eidx, const float* __restrict__ maskAtt,
                  const float* __restrict__ W1, const float* __restrict__ b1,
                  const float* __restrict__ W2, const float* __restrict__ b2,
                  const float* __restrict__ W3, const float* __restrict__ b3,
                  float* __restrict__ hPre) {
  extern __shared__ __align__(16) char smem[];
  __bf16* sW1  = (__bf16*)(smem + K1_OFF_W1T);
  __bf16* sW2  = (__bf16*)(smem + K1_OFF_W2T);
  __bf16* sW3  = (__bf16*)(smem + K1_OFF_W3T);
  __bf16* sA0  = (__bf16*)(smem + K1_OFF_A0);
  __bf16* sMid = (__bf16*)(smem + K1_OFF_MID);
  __bf16* sMd2 = (__bf16*)(smem + K1_OFF_MID2);
  int*    sIdx = (int*)   (smem + K1_OFF_IDX);
  float*  sMsk = (float*) (smem + K1_OFF_MSK);
  float*  sStg = (float*) (smem + K1_OFF_STG);

  const int tid = threadIdx.x;
  // stage weights transposed into LDS as [n][k] bf16 (coalesced global reads)
  for (int e = tid; e < HCc * Hc; e += 256)
    sW1[(e & 127) * 256 + (e >> 7)] = (__bf16)W1[e];
  for (int e = tid; e < Hc * Hc; e += 256)
    sW2[(e & 127) * 128 + (e >> 7)] = (__bf16)W2[e];
  for (int e = tid; e < Hc * Hc; e += 256)
    sW3[(e & 127) * 128 + (e >> 7)] = (__bf16)W3[e];

  const int lane  = tid & 31;
  const int w     = tid >> 5;          // 8 waves; wave owns 16-col slab
  const int lrow  = lane & 15;
  const int lhalf = lane >> 4;
  const int n0    = w * 16;
  const f4* hV4 = (const f4*)hV;
  const f4* sStg4 = (const f4*)sStg;

  for (int p = blockIdx.x; p < NPAIR; p += gridDim.x) {
    __syncthreads();                       // protect LDS from previous iter
    const int b = p >> 11;                 // N = 2048

    // kick the Tensor Data Mover: h_E[p] tile (48x128 f32) -> LDS staging.
    // TDM ignores EXEC; `w==0` is wave-uniform so it issues exactly once.
    if (w == 0)
      tdm_load_f32_tile(hE + (size_t)p * Kc * CEc, sStg, Kc, CEc, CEc);

    if (tid < Kc) {
      sIdx[tid] = Eidx[p * Kc + tid];
      sMsk[tid] = maskAtt[p * Kc + tid];
    }
    __syncthreads();

    // ---- gather h_V neighbor rows -> left half of A0 (overlaps TDM) ----
    for (int e = tid; e < Kc * 32; e += 256) {
      int k = e >> 5, c4 = e & 31;
      f4 v = hV4[((size_t)(b * Nc + sIdx[k])) * 32 + c4];
      bf16x4 bv = {(__bf16)v[0], (__bf16)v[1], (__bf16)v[2], (__bf16)v[3]};
      *(bf16x4*)(sA0 + k * LDA0 + c4 * 4) = bv;
    }
    if (w == 0) __builtin_amdgcn_s_wait_tensorcnt(0);
    __syncthreads();                       // TDM tile + h_V half visible

    // ---- convert staged h_E f32 -> right half of A0 (bf16) ----
    for (int e = tid; e < Kc * 32; e += 256) {
      int k = e >> 5, c4 = e & 31;
      f4 u = sStg4[k * 32 + c4];
      bf16x4 bu = {(__bf16)u[0], (__bf16)u[1], (__bf16)u[2], (__bf16)u[3]};
      *(bf16x4*)(sA0 + k * LDA0 + 128 + c4 * 4) = bu;
    }
    __syncthreads();

    // prefetch next pair's h_E rows into L2 ahead of its TDM
    if (p + (int)gridDim.x < NPAIR)
      __builtin_prefetch(hE + (size_t)(p + gridDim.x) * Kc * CEc, 0, 1);

    // ---- layer 1: 48x256 @ 256x128, gelu ----
    {
      v8f acc[3] = {};
      v16bf bcur = ld_frag(sW1 + (n0 + lrow) * 256 + lhalf * 16, 8);
#pragma unroll
      for (int ks = 0; ks < 8; ++ks) {
        v16bf bnext = bcur;
        if (ks < 7)   // prefetch next k-step's B fragment
          bnext = ld_frag(sW1 + (n0 + lrow) * 256 + (ks + 1) * 32 + lhalf * 16, 8);
        v16bf af[3];
#pragma unroll
        for (int mt = 0; mt < 3; ++mt)
          af[mt] = ld_frag(sA0 + (mt * 16 + lrow) * LDA0 + ks * 32 + lhalf * 16, 8);
#pragma unroll
        for (int mt = 0; mt < 3; ++mt)
          acc[mt] = wmma_bf16(af[mt], bcur, acc[mt]);
        bcur = bnext;
      }
      float bias = b1[n0 + lrow];
#pragma unroll
      for (int mt = 0; mt < 3; ++mt)
#pragma unroll
        for (int r = 0; r < 8; ++r) {
          float x = gelu_tanh(acc[mt][r] + bias);
          sMid[(mt * 16 + r + 8 * lhalf) * LDM + n0 + lrow] = (__bf16)x;
        }
    }
    __syncthreads();

    // ---- layer 2: 48x128 @ 128x128, gelu ----
    {
      v8f acc[3] = {};
      v16bf bcur = ld_frag(sW2 + (n0 + lrow) * 128 + lhalf * 16, 8);
#pragma unroll
      for (int ks = 0; ks < 4; ++ks) {
        v16bf bnext = bcur;
        if (ks < 3)
          bnext = ld_frag(sW2 + (n0 + lrow) * 128 + (ks + 1) * 32 + lhalf * 16, 8);
        v16bf af[3];
#pragma unroll
        for (int mt = 0; mt < 3; ++mt)
          af[mt] = ld_frag(sMid + (mt * 16 + lrow) * LDM + ks * 32 + lhalf * 16, 8);
#pragma unroll
        for (int mt = 0; mt < 3; ++mt)
          acc[mt] = wmma_bf16(af[mt], bcur, acc[mt]);
        bcur = bnext;
      }
      float bias = b2[n0 + lrow];
#pragma unroll
      for (int mt = 0; mt < 3; ++mt)
#pragma unroll
        for (int r = 0; r < 8; ++r) {
          float x = gelu_tanh(acc[mt][r] + bias);
          sMd2[(mt * 16 + r + 8 * lhalf) * LDM + n0 + lrow] = (__bf16)x;
        }
    }
    __syncthreads();

    // ---- layer 3 + masked K-reduction + residual ----
    {
      v8f acc[3] = {};
      v16bf bcur = ld_frag(sW3 + (n0 + lrow) * 128 + lhalf * 16, 8);
#pragma unroll
      for (int ks = 0; ks < 4; ++ks) {
        v16bf bnext = bcur;
        if (ks < 3)
          bnext = ld_frag(sW3 + (n0 + lrow) * 128 + (ks + 1) * 32 + lhalf * 16, 8);
        v16bf af[3];
#pragma unroll
        for (int mt = 0; mt < 3; ++mt)
          af[mt] = ld_frag(sMd2 + (mt * 16 + lrow) * LDM + ks * 32 + lhalf * 16, 8);
#pragma unroll
        for (int mt = 0; mt < 3; ++mt)
          acc[mt] = wmma_bf16(af[mt], bcur, acc[mt]);
        bcur = bnext;
      }
      float bias = b3[n0 + lrow];
      float total = 0.0f;
#pragma unroll
      for (int mt = 0; mt < 3; ++mt) {
        float s = 0.0f;
#pragma unroll
        for (int r = 0; r < 8; ++r)
          s += (acc[mt][r] + bias) * sMsk[mt * 16 + r + 8 * lhalf];
        total += s;
      }
      total += __shfl_xor(total, 16, 32);   // add the other 8-row half
      if (lane < 16) {
        int c = n0 + lane;
        hPre[(size_t)p * Hc + c] = hV[(size_t)p * Hc + c] + total * INV_SCALE;
      }
    }
  }
}

// =====================================================================
// Kernel 2a: midff = gelu_exact(hPre @ W_in + b_in)  (bf16 workspace)
#define K2A_OFF_W  0                         // 512 x 128 bf16 = 131072
#define K2A_OFF_A  131072                    // 64 x 136 bf16  = 17408
#define K2A_SMEM   (K2A_OFF_A + 17408)
// =====================================================================
__global__ __launch_bounds__(256, 1)
void ffn_in(const float* __restrict__ hPre, const float* __restrict__ W_in,
            const float* __restrict__ b_in, void* __restrict__ midff_raw) {
  extern __shared__ __align__(16) char smem[];
  __bf16* sW = (__bf16*)(smem + K2A_OFF_W);
  __bf16* sA = (__bf16*)(smem + K2A_OFF_A);
  __bf16* midff = (__bf16*)midff_raw;

  const int tid = threadIdx.x;
  for (int e = tid; e < Hc * FFc; e += 256)        // W_in[k][n] -> sW[n][k]
    sW[(e & 511) * 128 + (e >> 9)] = (__bf16)W_in[e];

  const int r0 = blockIdx.x * 64;
  const f4* h4 = (const f4*)hPre;
  for (int e = tid; e < 64 * 32; e += 256) {
    int row = e >> 5, c4 = e & 31;
    f4 v = h4[((size_t)(r0 + row)) * 32 + c4];
    bf16x4 bv = {(__bf16)v[0], (__bf16)v[1], (__bf16)v[2], (__bf16)v[3]};
    *(bf16x4*)(sA + row * 136 + c4 * 4) = bv;
  }
  __syncthreads();

  const int lane = tid & 31, w = tid >> 5;
  const int lrow = lane & 15, lhalf = lane >> 4;
  for (int ct = 0; ct < 4; ++ct) {
    const int n0 = (w * 4 + ct) * 16;
    v8f acc[4] = {};
    v16bf bcur = ld_frag(sW + (n0 + lrow) * 128 + lhalf * 16, 8);
#pragma unroll
    for (int ks = 0; ks < 4; ++ks) {
      v16bf bnext = bcur;
      if (ks < 3)
        bnext = ld_frag(sW + (n0 + lrow) * 128 + (ks + 1) * 32 + lhalf * 16, 8);
      v16bf af[4];
#pragma unroll
      for (int mt = 0; mt < 4; ++mt)
        af[mt] = ld_frag(sA + (mt * 16 + lrow) * 136 + ks * 32 + lhalf * 16, 8);
#pragma unroll
      for (int mt = 0; mt < 4; ++mt)
        acc[mt] = wmma_bf16(af[mt], bcur, acc[mt]);
      bcur = bnext;
    }
    float bias = b_in[n0 + lrow];
#pragma unroll
    for (int mt = 0; mt < 4; ++mt)
#pragma unroll
      for (int r = 0; r < 8; ++r) {
        float x = gelu_erf(acc[mt][r] + bias);
        midff[(size_t)(r0 + mt * 16 + r + 8 * lhalf) * FFc + n0 + lrow] = (__bf16)x;
      }
  }
}

// =====================================================================
// Kernel 2b: out = mask_V * (hPre + midff @ W_out + b_out)
#define K2B_OFF_W  0                         // 128 x 512 bf16 = 131072
#define K2B_OFF_A  131072                    // 64 x 520 bf16  = 66560
#define K2B_SMEM   (K2B_OFF_A + 66560)
// =====================================================================
__global__ __launch_bounds__(256, 1)
void ffn_out(const void* __restrict__ midff_raw, const float* __restrict__ hPre,
             const float* __restrict__ W_out, const float* __restrict__ b_out,
             const float* __restrict__ maskV, float* __restrict__ out) {
  extern __shared__ __align__(16) char smem[];
  __bf16* sW = (__bf16*)(smem + K2B_OFF_W);
  __bf16* sA = (__bf16*)(smem + K2B_OFF_A);
  const __bf16* midff = (const __bf16*)midff_raw;

  const int tid = threadIdx.x;
  for (int e = tid; e < FFc * Hc; e += 256)        // W_out[k][n] -> sW[n][k]
    sW[(e & 127) * 512 + (e >> 7)] = (__bf16)W_out[e];

  const int r0 = blockIdx.x * 64;
  const uint4* src = (const uint4*)(midff + (size_t)r0 * FFc);
  for (int e = tid; e < 64 * 64; e += 256) {       // 16B (8 bf16) chunks
    int row = e >> 6, c8 = e & 63;
    *(uint4*)(sA + row * 520 + c8 * 8) = src[row * 64 + c8];
  }
  __syncthreads();

  const int lane = tid & 31, w = tid >> 5;
  const int lrow = lane & 15, lhalf = lane >> 4;
  const int n0 = w * 16;
  v8f acc[4] = {};
  v16bf bcur = ld_frag(sW + (n0 + lrow) * 512 + lhalf * 16, 8);
#pragma unroll
  for (int ks = 0; ks < 16; ++ks) {
    v16bf bnext = bcur;
    if (ks < 15)
      bnext = ld_frag(sW + (n0 + lrow) * 512 + (ks + 1) * 32 + lhalf * 16, 8);
    v16bf af[4];
#pragma unroll
    for (int mt = 0; mt < 4; ++mt)
      af[mt] = ld_frag(sA + (mt * 16 + lrow) * 520 + ks * 32 + lhalf * 16, 8);
#pragma unroll
    for (int mt = 0; mt < 4; ++mt)
      acc[mt] = wmma_bf16(af[mt], bcur, acc[mt]);
    bcur = bnext;
  }
  float bias = b_out[n0 + lrow];
#pragma unroll
  for (int mt = 0; mt < 4; ++mt)
#pragma unroll
    for (int r = 0; r < 8; ++r) {
      int row = r0 + mt * 16 + r + 8 * lhalf;
      int c = n0 + lrow;
      float val = acc[mt][r] + bias + hPre[(size_t)row * Hc + c];
      out[(size_t)row * Hc + c] = maskV[row] * val;
    }
}

// =====================================================================
// host launcher
// inputs: 0 h_V, 1 h_E, 2 E_idx, 3 mask_V, 4 mask_attend,
//         5 W1, 6 b1, 7 W2, 8 b2, 9 W3, 10 b3, 11 W_in, 12 b_in,
//         13 W_out, 14 b_out
// =====================================================================
extern "C" void kernel_launch(void* const* d_in, const int* in_sizes, int n_in,
                              void* d_out, int out_size, void* d_ws, size_t ws_size,
                              hipStream_t stream) {
  const float* hV    = (const float*)d_in[0];
  const float* hE    = (const float*)d_in[1];
  const int*   Eidx  = (const int*)  d_in[2];
  const float* maskV = (const float*)d_in[3];
  const float* maskA = (const float*)d_in[4];
  const float* W1 = (const float*)d_in[5];
  const float* b1 = (const float*)d_in[6];
  const float* W2 = (const float*)d_in[7];
  const float* b2 = (const float*)d_in[8];
  const float* W3 = (const float*)d_in[9];
  const float* b3 = (const float*)d_in[10];
  const float* W_in  = (const float*)d_in[11];
  const float* b_in  = (const float*)d_in[12];
  const float* W_out = (const float*)d_in[13];
  const float* b_out = (const float*)d_in[14];
  float* out = (float*)d_out;

  // workspace: hPre f32 (4 MB) | midff bf16 (8 MB)
  float* hPre = (float*)d_ws;
  void*  midff = (void*)((char*)d_ws + (size_t)NPAIR * Hc * sizeof(float));

  enc_edge_mlp<<<1024, 256, K1_SMEM, stream>>>(hV, hE, Eidx, maskA,
                                               W1, b1, W2, b2, W3, b3, hPre);
  ffn_in<<<NPAIR / 64, 256, K2A_SMEM, stream>>>(hPre, W_in, b_in, midff);
  ffn_out<<<NPAIR / 64, 256, K2B_SMEM, stream>>>(midff, hPre, W_out, b_out,
                                                 maskV, out);
  (void)in_sizes; (void)n_in; (void)out_size; (void)ws_size;
}
